// ParallelDroplessMoE_27711128993794
// MI455X (gfx1250) — compile-verified, matrix operand order checked by reading, not run on hardware
//
#include <hip/hip_runtime.h>
#include <math.h>

// ---------------- Problem constants (match setup_inputs) ----------------
constexpr int S_ = 2048, B_ = 2, H_ = 2048, F_ = 8192, E_ = 8, TOPK_ = 2;
constexpr int T_ = S_ * B_;            // 4096 tokens
constexpr int NSLOT = T_ * TOPK_;      // 8192 routed slots
constexpr int BM = 128, BN = 128, BK = 32;
constexpr int MAX_TILES = NSLOT / BM + E_;   // 72
constexpr int ROWS_PAD = NSLOT + BM;         // 8320 (padded rows for safe tile loads)

// ---------------- Workspace layout (bytes) ----------------
constexpr size_t EI_OFF   = 0;                                  // int[NSLOT]
constexpr size_t EW_OFF   = 32768;                              // float[NSLOT]
constexpr size_t POS_OFF  = 65536;                              // int[NSLOT]  slot -> sorted pos
constexpr size_t SS_OFF   = 98304;                              // int[NSLOT]  sorted pos -> slot
constexpr size_t META_OFF = 131072;                             // ints: see below
constexpr size_t XG_OFF   = 262144;                             // bf16 [ROWS_PAD][H]
constexpr size_t XG_BYTES = (size_t)ROWS_PAD * H_ * 2;
constexpr size_t HBUF_OFF = XG_OFF + XG_BYTES;                  // bf16 [ROWS_PAD][F]
constexpr size_t HBUF_BYTES = (size_t)ROWS_PAD * F_ * 2;
constexpr size_t Y_OFF    = HBUF_OFF + HBUF_BYTES;              // float [NSLOT][H]
// meta ints: [0]=tileCount, [8..79]=tileExpert, [88..159]=tileM0, [168..239]=tileMend

typedef __bf16 v16bf __attribute__((ext_vector_type(16)));
typedef __bf16 v2bf  __attribute__((ext_vector_type(2)));
typedef float  v8f   __attribute__((ext_vector_type(8)));
typedef unsigned int v8u __attribute__((ext_vector_type(8)));

// Native bf16 conversion (backend emits v_cvt_*bf16* on gfx1250).
static __device__ __forceinline__ __bf16 f2bf(float f) { return (__bf16)f; }

// Pack two floats into one dword of bf16 pair -> single v_cvt_pk_bf16_f32.
static __device__ __forceinline__ unsigned pk2bf(float lo, float hi) {
    v2bf p; p[0] = (__bf16)lo; p[1] = (__bf16)hi;
    return __builtin_bit_cast(unsigned, p);
}

// Low 32 bits of a generic shared-aperture address == LDS byte offset on AMDGCN.
static __device__ __forceinline__ unsigned lds_off32(const void* p) {
    return (unsigned)(unsigned long long)p;
}

// Async raw copy global(16B/lane) -> LDS, tracked by ASYNCcnt.
static __device__ __forceinline__ void async_copy_b128(unsigned lds_dst,
                                                       unsigned long long gaddr) {
    asm volatile("global_load_async_to_lds_b128 %0, %1, off"
                 :: "v"(lds_dst), "v"(gaddr) : "memory");
}
static __device__ __forceinline__ void wait_async0() {
    asm volatile("s_wait_asynccnt 0" ::: "memory");
}

// ---------------- 1) Router: logits, softmax, top-2 ----------------
__global__ __launch_bounds__(256) void router_kernel(
    const float* __restrict__ x, const float* __restrict__ rw,
    int* __restrict__ ei, float* __restrict__ ew) {
    const int lane = threadIdx.x & 31;
    const int wid  = threadIdx.x >> 5;
    const int t    = blockIdx.x * 8 + wid;           // one wave per token
    float acc[E_];
#pragma unroll
    for (int e = 0; e < E_; ++e) acc[e] = 0.f;
    const float* xr = x + (size_t)t * H_;
    for (int h = lane; h < H_; h += 32) {
        float xv = xr[h];
#pragma unroll
        for (int e = 0; e < E_; ++e) acc[e] += xv * rw[h * E_ + e];
    }
#pragma unroll
    for (int off = 16; off > 0; off >>= 1) {
#pragma unroll
        for (int e = 0; e < E_; ++e) acc[e] += __shfl_xor(acc[e], off, 32);
    }
    if (lane == 0) {
        float m = acc[0];
#pragma unroll
        for (int e = 1; e < E_; ++e) m = fmaxf(m, acc[e]);
        float p[E_], sum = 0.f;
#pragma unroll
        for (int e = 0; e < E_; ++e) { p[e] = expf(acc[e] - m); sum += p[e]; }
        float inv = 1.f / sum;
#pragma unroll
        for (int e = 0; e < E_; ++e) p[e] *= inv;
        int i1 = 0;
#pragma unroll
        for (int e = 1; e < E_; ++e) if (p[e] > p[i1]) i1 = e;     // first index wins ties
        int i2 = (i1 == 0) ? 1 : 0;
#pragma unroll
        for (int e = 0; e < E_; ++e) if (e != i1 && p[e] > p[i2]) i2 = e;
        ei[2 * t]     = i1; ew[2 * t]     = p[i1];
        ei[2 * t + 1] = i2; ew[2 * t + 1] = p[i2];
    }
}

// ---------------- 2) Stable counting sort + ragged tile schedule ----------------
__global__ __launch_bounds__(256) void sort_schedule_kernel(
    const int* __restrict__ ei, int* __restrict__ pos,
    int* __restrict__ sortedSlot, int* __restrict__ meta) {
    __shared__ int s_cnt[256][E_];
    __shared__ int s_offs[E_ + 1];
    const int tid = threadIdx.x;
    const int base = tid * (NSLOT / 256);            // 32 slots / thread
    int c[E_];
#pragma unroll
    for (int e = 0; e < E_; ++e) c[e] = 0;
    for (int i = 0; i < NSLOT / 256; ++i) c[ei[base + i]]++;
#pragma unroll
    for (int e = 0; e < E_; ++e) s_cnt[tid][e] = c[e];
    __syncthreads();
    if (tid == 0) {
        int counts[E_];
        for (int e = 0; e < E_; ++e) {               // exclusive scan across threads
            int run = 0;
            for (int t = 0; t < 256; ++t) { int v = s_cnt[t][e]; s_cnt[t][e] = run; run += v; }
            counts[e] = run;
        }
        s_offs[0] = 0;
        for (int e = 0; e < E_; ++e) s_offs[e + 1] = s_offs[e] + counts[e];
        // ragged M-tile table
        int nt = 0;
        for (int e = 0; e < E_; ++e) {
            for (int m0 = s_offs[e]; m0 < s_offs[e + 1]; m0 += BM) {
                meta[8 + nt]   = e;
                meta[88 + nt]  = m0;
                meta[168 + nt] = min(m0 + BM, s_offs[e + 1]);
                nt++;
            }
        }
        meta[0] = nt;
    }
    __syncthreads();
    int r[E_];
#pragma unroll
    for (int e = 0; e < E_; ++e) r[e] = 0;
    for (int i = 0; i < NSLOT / 256; ++i) {
        int slot = base + i;
        int e = ei[slot];
        int p = s_offs[e] + s_cnt[tid][e] + r[e]; r[e]++;
        pos[slot] = p;
        sortedSlot[p] = slot;
    }
}

// ---------------- 3) Gather + fp32->bf16 (packed), zero-padded tail ----------------
__global__ __launch_bounds__(256) void gather_kernel(
    const float* __restrict__ x, const int* __restrict__ sortedSlot,
    __bf16* __restrict__ xg) {
    const int p = blockIdx.x;                         // 0 .. ROWS_PAD-1
    __bf16* dst = xg + (size_t)p * H_;
    const int c = threadIdx.x * 8;                    // 8 elements / thread (H=2048)
    if (p < NSLOT) {
        const int tok = sortedSlot[p] >> 1;
        const float* src = x + (size_t)tok * H_ + c;
        float4 a = *(const float4*)(src);
        float4 b = *(const float4*)(src + 4);
        uint4 u;
        u.x = pk2bf(a.x, a.y); u.y = pk2bf(a.z, a.w);
        u.z = pk2bf(b.x, b.y); u.w = pk2bf(b.z, b.w);
        *(uint4*)(dst + c) = u;
    } else {
        uint4 z = {0u, 0u, 0u, 0u};
        *(uint4*)(dst + c) = z;
    }
}

// ---------------- 4) Ragged grouped GEMM with bf16 WMMA ----------------
// A: bf16 [ROWS_PAD][K] (sorted activations) staged via async global->LDS copy,
// W: fp32 [E][K][N] converted to bf16 (packed pairs) while transposing into LDS,
// C: bf16 (gelu path) or fp32. 8 waves, 128x128x32 tile, wave = 4x2 of 16x16x32 WMMA.
template <bool GELU_BF16_OUT>
__global__ __launch_bounds__(256) void moe_gemm_kernel(
    const __bf16* __restrict__ A, const float* __restrict__ W,
    void* __restrict__ C, const int* __restrict__ meta, int K, int N) {
    __shared__ __align__(16) __bf16 sA[BM * BK];      // [128 rows][32 k]
    __shared__ __align__(16) __bf16 sB[BN * BK];      // transposed: [128 n][32 k]

    if (blockIdx.x >= meta[0]) return;
    const int e    = meta[8 + blockIdx.x];
    const int m0   = meta[88 + blockIdx.x];
    const int mend = meta[168 + blockIdx.x];
    const int n0   = blockIdx.y * BN;
    const float* Wp = W + (size_t)e * K * N;

    const int tid = threadIdx.x, lane = tid & 31, wid = tid >> 5;
    const int wm = wid >> 2, wn = wid & 3;            // 2x4 wave grid
    const int half = lane >> 4, l16 = lane & 15;

    // A tile: 2 x 16B async copies per thread per K-step (512 chunks, 4/row).
    const int r0 = tid >> 2,         c0 = tid & 3;
    const int r1 = (tid + 256) >> 2, c1 = (tid + 256) & 3;
    const unsigned ldsA0 = lds_off32(sA + r0 * BK + c0 * 8);
    const unsigned ldsA1 = lds_off32(sA + r1 * BK + c1 * 8);
    const __bf16* ga0 = A + (size_t)(m0 + r0) * K + c0 * 8;
    const __bf16* ga1 = A + (size_t)(m0 + r1) * K + c1 * 8;

    // B tile: thread g handles a K-pair x 4 N columns; 2 groups/thread.
    // g in [0,512): k2 = g>>5 (K rows 2*k2, 2*k2+1), n4 = (g&31)*4.
    const int k2a = tid >> 5,          n4a = (tid & 31) * 4;
    const int k2b = (tid + 256) >> 5,  n4b = (tid & 31) * 4;   // (tid+256)&31 == tid&31
    const float* wpa0 = Wp + (size_t)(2 * k2a) * N + n0 + n4a;
    const float* wpa1 = wpa0 + N;
    const float* wpb0 = Wp + (size_t)(2 * k2b) * N + n0 + n4b;
    const float* wpb1 = wpb0 + N;
    unsigned* sba = (unsigned*)(sB + n4a * BK + 2 * k2a);   // +j rows => +32*BK bytes? no: elements
    unsigned* sbb = (unsigned*)(sB + n4b * BK + 2 * k2b);
    const size_t wstep = (size_t)BK * N;              // advance BK K-rows per step

    v8f acc[4][2] = {};

    for (int kb = 0; kb < K; kb += BK) {
        // A tile: async raw copy (bf16 already).
        async_copy_b128(ldsA0, (unsigned long long)(ga0 + kb));
        async_copy_b128(ldsA1, (unsigned long long)(ga1 + kb));

        // B tile: two K-adjacent float4s -> 4 packed bf16 pairs -> 4 b32 LDS stores.
        {
            float4 a0 = *(const float4*)wpa0;
            float4 a1 = *(const float4*)wpa1;
            float4 b0 = *(const float4*)wpb0;
            float4 b1 = *(const float4*)wpb1;
            // sB rows are BK bf16 = 64B = 16 dwords apart.
            sba[0 * (BK / 2)] = pk2bf(a0.x, a1.x);
            sba[1 * (BK / 2)] = pk2bf(a0.y, a1.y);
            sba[2 * (BK / 2)] = pk2bf(a0.z, a1.z);
            sba[3 * (BK / 2)] = pk2bf(a0.w, a1.w);
            sbb[0 * (BK / 2)] = pk2bf(b0.x, b1.x);
            sbb[1 * (BK / 2)] = pk2bf(b0.y, b1.y);
            sbb[2 * (BK / 2)] = pk2bf(b0.z, b1.z);
            sbb[3 * (BK / 2)] = pk2bf(b0.w, b1.w);
            wpa0 += wstep; wpa1 += wstep; wpb0 += wstep; wpb1 += wstep;
        }
        if (kb + BK < K)  // stream next weight tile toward L2
            __builtin_prefetch(wpa0, 0, 1);
        wait_async0();
        __syncthreads();

        // Per-lane fragments per ISA 16-bit layouts:
        // lane<16 holds K {0..7,16..23}; lane>=16 holds K {8..15,24..31}.
        v16bf bfrag[2];
#pragma unroll
        for (int n = 0; n < 2; ++n) {
            const __bf16* p = sB + (wn * 32 + n * 16 + l16) * BK + half * 8;
            uint4 lo = *(const uint4*)p;
            uint4 hi = *(const uint4*)(p + 16);
            v8u tmp;
            tmp[0] = lo.x; tmp[1] = lo.y; tmp[2] = lo.z; tmp[3] = lo.w;
            tmp[4] = hi.x; tmp[5] = hi.y; tmp[6] = hi.z; tmp[7] = hi.w;
            bfrag[n] = __builtin_bit_cast(v16bf, tmp);
        }
#pragma unroll
        for (int m = 0; m < 4; ++m) {
            const __bf16* p = sA + (wm * 64 + m * 16 + l16) * BK + half * 8;
            uint4 lo = *(const uint4*)p;
            uint4 hi = *(const uint4*)(p + 16);
            v8u tmp;
            tmp[0] = lo.x; tmp[1] = lo.y; tmp[2] = lo.z; tmp[3] = lo.w;
            tmp[4] = hi.x; tmp[5] = hi.y; tmp[6] = hi.z; tmp[7] = hi.w;
            v16bf afrag = __builtin_bit_cast(v16bf, tmp);
#pragma unroll
            for (int n = 0; n < 2; ++n)
                acc[m][n] = __builtin_amdgcn_wmma_f32_16x16x32_bf16(
                    false, afrag, false, bfrag[n], (short)0, acc[m][n], false, false);
        }
        __syncthreads();
    }

    // Epilogue; ragged rows masked at the store only (WMMA ran with full EXEC).
#pragma unroll
    for (int m = 0; m < 4; ++m) {
#pragma unroll
        for (int n = 0; n < 2; ++n) {
            int col = n0 + wn * 32 + n * 16 + l16;
#pragma unroll
            for (int r = 0; r < 8; ++r) {
                int row = m0 + wm * 64 + m * 16 + half * 8 + r;
                if (row < mend) {
                    float v = acc[m][n][r];
                    if (GELU_BF16_OUT) {
                        v = 0.5f * v * (1.f + erff(v * 0.70710678118654752f));
                        ((__bf16*)C)[(size_t)row * N + col] = f2bf(v);
                    } else {
                        ((float*)C)[(size_t)row * N + col] = v;
                    }
                }
            }
        }
    }
}

// ---------------- 5) Combine: out[t] = ew0*y[pos(2t)] + ew1*y[pos(2t+1)] ----------------
__global__ __launch_bounds__(256) void combine_kernel(
    const float* __restrict__ y, const int* __restrict__ pos,
    const float* __restrict__ ew, float* __restrict__ out) {
    const int t = blockIdx.x;
    const float* y0 = y + (size_t)pos[2 * t] * H_;
    const float* y1 = y + (size_t)pos[2 * t + 1] * H_;
    const float w0 = ew[2 * t], w1 = ew[2 * t + 1];
    float* o = out + (size_t)t * H_;
    for (int c = threadIdx.x; c < H_; c += 256)
        o[c] = w0 * y0[c] + w1 * y1[c];
}

// ---------------- 6) Buffer: per-(token,expert) raw y row or zeros ----------------
__global__ __launch_bounds__(256) void buffer_kernel(
    const float* __restrict__ y, const int* __restrict__ ei,
    const int* __restrict__ pos, float* __restrict__ buffer) {
    const int b = blockIdx.x;            // 0 .. T*E-1
    const int t = b >> 3, e = b & 7;
    const float* src = nullptr;
    if (e == ei[2 * t])          src = y + (size_t)pos[2 * t] * H_;
    else if (e == ei[2 * t + 1]) src = y + (size_t)pos[2 * t + 1] * H_;
    float* dst = buffer + (size_t)b * H_;
    if (src) { for (int c = threadIdx.x; c < H_; c += 256) dst[c] = src[c]; }
    else     { for (int c = threadIdx.x; c < H_; c += 256) dst[c] = 0.f; }
}

// ---------------- Launch ----------------
extern "C" void kernel_launch(void* const* d_in, const int* in_sizes, int n_in,
                              void* d_out, int out_size, void* d_ws, size_t ws_size,
                              hipStream_t stream) {
    const float* x  = (const float*)d_in[0];
    const float* rw = (const float*)d_in[1];
    const float* w1 = (const float*)d_in[2];
    const float* w2 = (const float*)d_in[3];

    char* ws = (char*)d_ws;
    int*    ei  = (int*)(ws + EI_OFF);
    float*  ew  = (float*)(ws + EW_OFF);
    int*    pos = (int*)(ws + POS_OFF);
    int*    ss  = (int*)(ws + SS_OFF);
    int*    meta = (int*)(ws + META_OFF);
    __bf16* xg  = (__bf16*)(ws + XG_OFF);
    __bf16* hb  = (__bf16*)(ws + HBUF_OFF);
    float*  y   = (float*)(ws + Y_OFF);

    float* out    = (float*)d_out;                    // [T][H]
    float* buffer = out + (size_t)T_ * H_;            // [T][E][H]

    router_kernel<<<T_ / 8, 256, 0, stream>>>(x, rw, ei, ew);
    sort_schedule_kernel<<<1, 256, 0, stream>>>(ei, pos, ss, meta);
    gather_kernel<<<ROWS_PAD, 256, 0, stream>>>(x, ss, xg);
    moe_gemm_kernel<true><<<dim3(MAX_TILES, F_ / BN), 256, 0, stream>>>(
        xg, w1, (void*)hb, meta, /*K=*/H_, /*N=*/F_);
    moe_gemm_kernel<false><<<dim3(MAX_TILES, H_ / BN), 256, 0, stream>>>(
        hb, w2, (void*)y, meta, /*K=*/F_, /*N=*/H_);
    combine_kernel<<<T_, 256, 0, stream>>>(y, pos, ew, out);
    buffer_kernel<<<T_ * E_, 256, 0, stream>>>(y, ei, pos, buffer);
    (void)in_sizes; (void)n_in; (void)out_size; (void)ws_size;
}
